// CWCFace_11201274708637
// MI455X (gfx1250) — matrix operand order repaired and checked by reading, not compiled, not run on hardware
//
#include <hip/hip_runtime.h>
#include <cstddef>

#define EMB     512
#define BATCH   512
#define CLS     70722
#define MARGIN  0.4f
#define HCOEF   0.333f
#define SSCALE  64.0f
#define EPSF    0.001f
#define PI_F    3.14159265358979f

#define MCHUNKS (BATCH / 16)        // 32
#define KCHUNKS (EMB / 32)          // 16
#define NFRAGS  (MCHUNKS * KCHUNKS) // 512
#define WS_NEEDED ((size_t)NFRAGS * 2048)  // 1 MB

typedef __attribute__((ext_vector_type(16))) __bf16 v16bf;
typedef __attribute__((ext_vector_type(8)))  float  v8f;
typedef __attribute__((ext_vector_type(2)))  float  f2v;

union FragU {
    v16bf v;
    uint4 q[2];
    unsigned u[8];
};

// Pack hi-bf16 of two floats into one dword with v_perm_b32; lo residual pair
// with fma_mix-able subtract + perm. Truncation split: x == hi + r exactly.
__device__ __forceinline__ void split_pack_pair(float x0, float x1,
                                                unsigned &hp, unsigned &lp) {
    unsigned u0 = __float_as_uint(x0), u1 = __float_as_uint(x1);
    hp = __builtin_amdgcn_perm(u1, u0, 0x07060302u);
    float r0 = x0 - __uint_as_float(u0 & 0xffff0000u);
    float r1 = x1 - __uint_as_float(u1 & 0xffff0000u);
    lp = __builtin_amdgcn_perm(__float_as_uint(r1), __float_as_uint(r0), 0x07060302u);
}

// ---------------------------------------------------------------------------
// Pre-pack embeddings (A) into fragment-major bf16 hi/lo in d_ws (1 MB).
// Frag f = mchunk*KCHUNKS + kchunk; per lane 16 bf16 hi at f*2048+lane*32,
// lo at +1024. A 16x32 bf16 layout: lane(lr,hgrp): row=m0+lr,
// elems 0..15 = K k0+hgrp*16+e.
// ---------------------------------------------------------------------------
__global__ __launch_bounds__(256) void cwcface_prepack_a(
    const float* __restrict__ A, uint4* __restrict__ Apk)
{
    const int lane = threadIdx.x & 31;
    const int wave = threadIdx.x >> 5;
    const int f    = blockIdx.x * 8 + wave;      // 0..511
    const int mchunk = f >> 4;
    const int kchunk = f & (KCHUNKS - 1);
    const int hgrp = lane >> 4;
    const int lr   = lane & 15;

    const float* ap = A + (size_t)(mchunk * 16 + lr) * EMB + kchunk * 32 + hgrp * 16;
    float x[16];
    #pragma unroll
    for (int e = 0; e < 8; ++e) { x[e] = ap[e]; x[8 + e] = ap[8 + e]; }

    FragU h, l;
    #pragma unroll
    for (int e = 0; e < 16; e += 2)
        split_pack_pair(x[e], x[e + 1], h.u[e >> 1], l.u[e >> 1]);

    const int qbase = f * 128 + lane * 2;
    Apk[qbase]      = h.q[0];
    Apk[qbase + 1]  = h.q[1];
    Apk[qbase + 64] = l.q[0];
    Apk[qbase + 65] = l.q[1];
}

// Cooperative W-chunk loaders: thread t owns krow = t>>3, cols cbase..cbase+3.
__device__ __forceinline__ void load_wchunk(const float* __restrict__ W,
                                            int kbase, int krow, int n0,
                                            int cbase, bool edge, float x[4]) {
    if (!edge) {
        const f2v* p = (const f2v*)(W + (size_t)(kbase + krow) * CLS + n0 + cbase);
        f2v a = __builtin_nontemporal_load(p);
        f2v b = __builtin_nontemporal_load(p + 1);
        x[0] = a[0]; x[1] = a[1]; x[2] = b[0]; x[3] = b[1];
    } else {
        #pragma unroll
        for (int j = 0; j < 4; ++j) {
            int cc = n0 + cbase + j; cc = (cc < CLS) ? cc : (CLS - 1);
            x[j] = W[(size_t)(kbase + krow) * CLS + cc];
        }
    }
}

// Store chunk into LDS in B-frag layout ([col*32 + krow], hi & lo planes).
// hi-bf16 is just the float's top 16 bits; lo needs one residual op.
__device__ __forceinline__ void store_wchunk(unsigned short* __restrict__ shi,
                                             unsigned short* __restrict__ slo,
                                             int krow, int cbase,
                                             const float x[4], float ss[4]) {
    #pragma unroll
    for (int j = 0; j < 4; ++j) {
        unsigned u = __float_as_uint(x[j]);
        int idx = (cbase + j) * 32 + krow;
        shi[idx] = (unsigned short)(u >> 16);
        float r = x[j] - __uint_as_float(u & 0xffff0000u);
        slo[idx] = (unsigned short)(__float_as_uint(r) >> 16);
        ss[j] = fmaf(x[j], x[j], ss[j]);
    }
}

__device__ __forceinline__ void read_frag(const unsigned short* __restrict__ plane,
                                          int lr, int hgrp, int g, FragU &f) {
    const uint4* q = (const uint4*)(plane + ((g * 16 + lr) * 32 + hgrp * 16));
    f.q[0] = q[0];
    f.q[1] = q[1];
}

// ---------------------------------------------------------------------------
// Staged GEMM: block = 256 threads (8 waves), tile 512 rows x 32 cols.
// W chunk (32x32) loaded once per block per k-step into LDS (bf16 hi/lo,
// frag layout, double buffered); A comes pre-packed from d_ws. Wave w:
// rows [w*64,+64) x 32 cols, 3-product bf16 split (hi*hi + hi*lo + lo*hi).
// ---------------------------------------------------------------------------
__global__ __launch_bounds__(256) void cwcface_gemm_staged(
    const float* __restrict__ W, const uint4* __restrict__ Apk,
    float* __restrict__ out)
{
    __shared__ __attribute__((aligned(16))) unsigned short sW[2][2][1024];
    __shared__ float colssq[32];

    const int t    = threadIdx.x;
    const int lane = t & 31;
    const int wave = t >> 5;
    const int hgrp = lane >> 4;
    const int lr   = lane & 15;
    const int n0   = blockIdx.x * 32;
    const bool edge = (n0 + 32 > CLS);

    const int krow  = t >> 3;          // 0..31
    const int cbase = (t & 7) * 4;     // 0,4,...,28

    if (t < 32) colssq[t] = 0.0f;

    float ss[4] = {0.f, 0.f, 0.f, 0.f};
    float cur[4];
    load_wchunk(W, 0, krow, n0, cbase, edge, cur);
    store_wchunk(&sW[0][0][0], &sW[0][1][0], krow, cbase, cur, ss);
    __syncthreads();

    v8f acc[4][2] = {};

    for (int k0 = 0; k0 < EMB; k0 += 32) {
        const int p = (k0 >> 5) & 1;
        const bool more = (k0 + 32 < EMB);

        float nxt[4];
        if (more) load_wchunk(W, k0 + 32, krow, n0, cbase, edge, nxt);

        FragU bh0, bl0, bh1, bl1;
        read_frag(&sW[p][0][0], lr, hgrp, 0, bh0);
        read_frag(&sW[p][1][0], lr, hgrp, 0, bl0);
        read_frag(&sW[p][0][0], lr, hgrp, 1, bh1);
        read_frag(&sW[p][1][0], lr, hgrp, 1, bl1);

        #pragma unroll
        for (int s = 0; s < 4; ++s) {
            const int f = (wave * 4 + s) * KCHUNKS + (k0 >> 5);
            const int qbase = f * 128 + lane * 2;
            FragU ah, al;
            ah.q[0] = Apk[qbase];      ah.q[1] = Apk[qbase + 1];
            al.q[0] = Apk[qbase + 64]; al.q[1] = Apk[qbase + 65];

            acc[s][0] = __builtin_amdgcn_wmma_f32_16x16x32_bf16(false, ah.v, false, bh0.v, (short)0, acc[s][0], false, false);
            acc[s][0] = __builtin_amdgcn_wmma_f32_16x16x32_bf16(false, ah.v, false, bl0.v, (short)0, acc[s][0], false, false);
            acc[s][0] = __builtin_amdgcn_wmma_f32_16x16x32_bf16(false, al.v, false, bh0.v, (short)0, acc[s][0], false, false);
            acc[s][1] = __builtin_amdgcn_wmma_f32_16x16x32_bf16(false, ah.v, false, bh1.v, (short)0, acc[s][1], false, false);
            acc[s][1] = __builtin_amdgcn_wmma_f32_16x16x32_bf16(false, ah.v, false, bl1.v, (short)0, acc[s][1], false, false);
            acc[s][1] = __builtin_amdgcn_wmma_f32_16x16x32_bf16(false, al.v, false, bh1.v, (short)0, acc[s][1], false, false);
        }

        if (more) {
            store_wchunk(&sW[1 - p][0][0], &sW[1 - p][1][0], krow, cbase, nxt, ss);
            __syncthreads();
        }
    }

    // Column sum-of-squares reduction (each thread owns 4 fixed columns).
    #pragma unroll
    for (int j = 0; j < 4; ++j) atomicAdd(&colssq[cbase + j], ss[j]);
    __syncthreads();

    const float inv0 = rsqrtf(colssq[lr]);
    const float inv1 = rsqrtf(colssq[16 + lr]);

    // Epilogue: cosine = clip(dot*invnorm, -1+eps, 1-eps); out = cosine*S.
    // (For non-target cols, cos(clip(acos(cosine))) == cosine exactly.)
    // C/D layout: lane(lr,hgrp), VGPR v -> row = v + 8*hgrp, col = lr.
    const int col0 = n0 + lr, col1 = n0 + 16 + lr;
    const int m0 = wave * 64;
    const float clo = -1.0f + EPSF, chi = 1.0f - EPSF;
    #pragma unroll
    for (int s = 0; s < 4; ++s) {
        const int rowb = m0 + s * 16 + hgrp * 8;
        #pragma unroll
        for (int v = 0; v < 8; ++v) {
            size_t rbase = (size_t)(rowb + v) * CLS;
            if (col0 < CLS) {
                float c = fminf(fmaxf(acc[s][0][v] * inv0, clo), chi);
                __builtin_nontemporal_store(c * SSCALE, &out[rbase + col0]);
            }
            if (col1 < CLS) {
                float c = fminf(fmaxf(acc[s][1][v] * inv1, clo), chi);
                __builtin_nontemporal_store(c * SSCALE, &out[rbase + col1]);
            }
        }
    }
}

// ---------------------------------------------------------------------------
// Fallback (no workspace): fully fused, per-wave on-the-fly conversion.
// ---------------------------------------------------------------------------
__global__ __launch_bounds__(256) void cwcface_gemm_nows(
    const float* __restrict__ A, const float* __restrict__ W,
    float* __restrict__ out)
{
    const int lane = threadIdx.x & 31;
    const int wave = threadIdx.x >> 5;
    const int hgrp = lane >> 4;
    const int lr   = lane & 15;
    const int n0   = blockIdx.x * 32;
    const int m0   = wave * 64;

    const int col0 = n0 + lr, col1 = n0 + 16 + lr;
    const int c0c  = (col0 < CLS) ? col0 : (CLS - 1);
    const int c1c  = (col1 < CLS) ? col1 : (CLS - 1);

    v8f acc[4][2] = {};
    float ssq0 = 0.0f, ssq1 = 0.0f;

    for (int k0 = 0; k0 < EMB; k0 += 32) {
        FragU bh0, bl0, bh1, bl1;
        #pragma unroll
        for (int e = 0; e < 16; e += 2) {
            size_t kr0 = (size_t)(k0 + hgrp * 16 + e) * CLS;
            size_t kr1 = kr0 + CLS;
            float x0 = W[kr0 + c0c], x1 = W[kr1 + c0c];
            float y0 = W[kr0 + c1c], y1 = W[kr1 + c1c];
            ssq0 = fmaf(x0, x0, ssq0); ssq0 = fmaf(x1, x1, ssq0);
            ssq1 = fmaf(y0, y0, ssq1); ssq1 = fmaf(y1, y1, ssq1);
            split_pack_pair(x0, x1, bh0.u[e >> 1], bl0.u[e >> 1]);
            split_pack_pair(y0, y1, bh1.u[e >> 1], bl1.u[e >> 1]);
        }
        #pragma unroll
        for (int s = 0; s < 4; ++s) {
            const float* ap = A + (size_t)(m0 + s * 16 + lr) * EMB + k0 + hgrp * 16;
            float x[16];
            #pragma unroll
            for (int e = 0; e < 8; ++e) { x[e] = ap[e]; x[8 + e] = ap[8 + e]; }
            FragU ah, al;
            #pragma unroll
            for (int e = 0; e < 16; e += 2)
                split_pack_pair(x[e], x[e + 1], ah.u[e >> 1], al.u[e >> 1]);
            acc[s][0] = __builtin_amdgcn_wmma_f32_16x16x32_bf16(false, ah.v, false, bh0.v, (short)0, acc[s][0], false, false);
            acc[s][0] = __builtin_amdgcn_wmma_f32_16x16x32_bf16(false, ah.v, false, bl0.v, (short)0, acc[s][0], false, false);
            acc[s][0] = __builtin_amdgcn_wmma_f32_16x16x32_bf16(false, al.v, false, bh0.v, (short)0, acc[s][0], false, false);
            acc[s][1] = __builtin_amdgcn_wmma_f32_16x16x32_bf16(false, ah.v, false, bh1.v, (short)0, acc[s][1], false, false);
            acc[s][1] = __builtin_amdgcn_wmma_f32_16x16x32_bf16(false, ah.v, false, bl1.v, (short)0, acc[s][1], false, false);
            acc[s][1] = __builtin_amdgcn_wmma_f32_16x16x32_bf16(false, al.v, false, bh1.v, (short)0, acc[s][1], false, false);
        }
    }

    ssq0 += __shfl_xor(ssq0, 16, 32);
    ssq1 += __shfl_xor(ssq1, 16, 32);
    const float inv0 = rsqrtf(ssq0);
    const float inv1 = rsqrtf(ssq1);

    const float clo = -1.0f + EPSF, chi = 1.0f - EPSF;
    #pragma unroll
    for (int s = 0; s < 4; ++s) {
        const int rowb = m0 + s * 16 + hgrp * 8;
        #pragma unroll
        for (int v = 0; v < 8; ++v) {
            size_t rbase = (size_t)(rowb + v) * CLS;
            if (col0 < CLS) {
                float c = fminf(fmaxf(acc[s][0][v] * inv0, clo), chi);
                out[rbase + col0] = c * SSCALE;
            }
            if (col1 < CLS) {
                float c = fminf(fmaxf(acc[s][1][v] * inv1, clo), chi);
                out[rbase + col1] = c * SSCALE;
            }
        }
    }
}

// One block, 512 threads: per-class norm stats (O(B^2) LDS scan) and
// margin rewrite of the 512 target elements.
__global__ __launch_bounds__(512) void cwcface_fixup(
    const float* __restrict__ norms, const int* __restrict__ label,
    float* __restrict__ out)
{
    __shared__ float sv[BATCH];
    __shared__ int   sl[BATCH];
    const int i = threadIdx.x;
    const float vi = fminf(fmaxf(norms[i], 0.001f), 100.0f);
    sv[i] = vi;
    sl[i] = label[i];
    __syncthreads();

    const int li = sl[i];
    float n = 0.0f, s = 0.0f, sq = 0.0f;
    for (int j = 0; j < BATCH; ++j) {
        if (sl[j] == li) {
            float vj = sv[j];
            n += 1.0f; s += vj; sq = fmaf(vj, vj, sq);
        }
    }
    const float mean = s / n;
    const float var  = (sq - n * mean * mean) / fmaxf(n - 1.0f, 1.0f);
    const float sd   = sqrtf(fmaxf(var, 0.0f));
    const float res  = (n > 2.0f) ? (vi - mean) / (sd + EPSF) : (vi - mean) * 0.05f;
    const float ms   = fminf(fmaxf(res * HCOEF, -1.0f), 1.0f);

    const size_t idx = (size_t)i * CLS + li;
    const float cosine  = out[idx] * (1.0f / SSCALE);
    const float theta   = acosf(cosine);
    const float theta_m = fminf(fmaxf(theta - MARGIN * ms, EPSF), PI_F - EPSF);
    const float g_add   = MARGIN + MARGIN * ms;
    out[idx] = (cosf(theta_m) - g_add) * SSCALE;
}

extern "C" void kernel_launch(void* const* d_in, const int* in_sizes, int n_in,
                              void* d_out, int out_size, void* d_ws, size_t ws_size,
                              hipStream_t stream) {
    (void)in_sizes; (void)n_in; (void)out_size;
    const float* emb   = (const float*)d_in[0];   // [512, 512] f32
    const float* norms = (const float*)d_in[1];   // [512, 1]   f32
    const int*   label = (const int*)d_in[2];     // [512]      i32
    const float* W     = (const float*)d_in[3];   // [512, 70722] f32
    float* out = (float*)d_out;                   // [512, 70722] f32

    const int nblocks = (CLS + 31) / 32;          // 2211 column-tile blocks

    if (d_ws != nullptr && ws_size >= WS_NEEDED) {
        uint4* Apk = (uint4*)d_ws;
        cwcface_prepack_a<<<dim3(NFRAGS / 8), dim3(256), 0, stream>>>(emb, Apk);
        cwcface_gemm_staged<<<dim3(nblocks), dim3(256), 0, stream>>>(W, Apk, out);
    } else {
        cwcface_gemm_nows<<<dim3(nblocks), dim3(256), 0, stream>>>(emb, W, out);
    }
    cwcface_fixup<<<dim3(1), dim3(512), 0, stream>>>(norms, label, out);
}
